// NEDB_89266600280279
// MI455X (gfx1250) — matrix operand order repaired and checked by reading, not compiled, not run on hardware
//
#include <hip/hip_runtime.h>

// ---------------------------------------------------------------------------
// CDNA5 (gfx1250) WMMA types / helpers
// ---------------------------------------------------------------------------
typedef __attribute__((ext_vector_type(16))) _Float16 v16h;
typedef __attribute__((ext_vector_type(8)))  float    v8f;

__device__ __forceinline__ v8f wmma32(v16h a, v16h b, v8f c) {
  // D = A(16x32 f16) * B(32x16 f16) + C(16x16 f32)
  return __builtin_amdgcn_wmma_f32_16x16x32_f16(
      /*neg_a=*/false, a, /*neg_b=*/false, b,
      /*c_mod=*/(short)0, c, /*reuse_a=*/false, /*reuse_b=*/false);
}

// A fragment (16x32 f16), contiguous per-lane halves (matches B frags):
//  lanes 0-15 : row M=lane,    halves 0-15 -> K=0..15
//  lanes 16-31: row M=lane-16, halves 0-15 -> K=16..31
__device__ __forceinline__ v16h lds_a16(const _Float16* tile, int ld, int lane) {
  const _Float16* p = tile + (lane & 15) * ld + ((lane & 16) ? 16 : 0);
  v16h a;
#pragma unroll
  for (int i = 0; i < 16; ++i) a[i] = p[i];
  return a;
}

// B fragment from fragment-ordered LDS tile [col][k] (k contiguous):
// caller passes p = tile + col*32 + (lane&16); 16 contiguous halfs -> 2x ds_load_b128
__device__ __forceinline__ v16h lds_frag_h(const _Float16* p) {
  v16h v;
#pragma unroll
  for (int i = 0; i < 16; ++i) v[i] = p[i];
  return v;
}

// Same, but source tile holds raw f32 (async-DMA'd); convert during frag build.
__device__ __forceinline__ v16h lds_frag_f32(const float* p) {
  v16h v;
#pragma unroll
  for (int i = 0; i < 16; ++i) v[i] = (_Float16)p[i];
  return v;
}

// Per-lane async global->LDS copy (4B per lane), GV addressing. ASYNCcnt-tracked.
__device__ __forceinline__ void async_ld_b32(unsigned lds_off, const float* g) {
  asm volatile("global_load_async_to_lds_b32 %0, %1, off"
               :: "v"(lds_off), "v"(g)
               : "memory");
}

// D fragment (16x16 f32): element r of v8f <-> row M = r + ((lane&16)>>1), col N = lane&15

// ---------------------------------------------------------------------------
// Generic 1x1-conv GEMM:  Out[b,m,n] = sum_k Wt[m,k] * In[b,k,n] + bias[m] (+Res)
// N = 4096 (H*W). Block = 128 threads (4 waves). Tile: M=32 (grid.y blocks), N=64.
// ---------------------------------------------------------------------------
__global__ void gemm1x1_wmma(const float* __restrict__ Wt, const float* __restrict__ In,
                             const float* __restrict__ bias, const float* __restrict__ Res,
                             float* __restrict__ Out,
                             int K, long long inb, long long resb, long long outb) {
  const int N = 4096;
  __shared__ alignas(16) _Float16 As[32 * 160];   // weights tile [m][k], K <= 160
  __shared__ alignas(16) _Float16 BsT[64 * 32];   // input chunk, fragment order [n][k]
  const int tid = threadIdx.x, lane = tid & 31, wv = tid >> 5;
  const int n0 = blockIdx.x * 64, mblk = blockIdx.y, b = blockIdx.z;

  for (int i = tid; i < 32 * K; i += 128) {
    int m = i / K, k = i - m * K;
    As[i] = (_Float16)Wt[(mblk * 32 + m) * K + k];
  }

  const float* inB = In + (long long)b * inb + n0;
  v8f acc0 = {}, acc1 = {};
  for (int k0 = 0; k0 < K; k0 += 32) {
    __syncthreads();
    // float4 global loads (coalesced along n), transposed f16 stores -> [n][k]
    for (int i = tid; i < 512; i += 128) {
      int kk = i >> 4, n4 = (i & 15) << 2;
      float4 v = *(const float4*)(inB + (long long)(k0 + kk) * N + n4);
      _Float16* d = BsT + n4 * 32 + kk;
      d[0]  = (_Float16)v.x;
      d[32] = (_Float16)v.y;
      d[64] = (_Float16)v.z;
      d[96] = (_Float16)v.w;
    }
    if (k0 + 32 < K)
      __builtin_prefetch(inB + (long long)(k0 + 32 + (tid >> 2)) * N + ((tid & 3) << 4), 0, 0);
    __syncthreads();
    v16h a0 = lds_a16(As + k0, K, lane);
    v16h a1 = lds_a16(As + 16 * K + k0, K, lane);
    v16h bf = lds_frag_h(BsT + (wv * 16 + (lane & 15)) * 32 + (lane & 16));
    acc0 = wmma32(a0, bf, acc0);
    acc1 = wmma32(a1, bf, acc1);
  }

  const int col = lane & 15, rh = (lane >> 4) & 1;
  const int n = n0 + wv * 16 + col;
#pragma unroll
  for (int r = 0; r < 8; ++r) {
    int m0 = mblk * 32 + r + rh * 8;
    int m1 = m0 + 16;
    float v0 = acc0[r] + bias[m0];
    float v1 = acc1[r] + bias[m1];
    if (Res) {
      v0 += Res[(long long)b * resb + (long long)m0 * N + n];
      v1 += Res[(long long)b * resb + (long long)m1 * N + n];
    }
    Out[(long long)b * outb + (long long)m0 * N + n] = v0;
    Out[(long long)b * outb + (long long)m1 * N + n] = v1;
  }
}

// ---------------------------------------------------------------------------
// 2x2 max-pool:  [B,32,64,64] -> [B,32,32,32]
// ---------------------------------------------------------------------------
__global__ void maxpool2_k(const float* __restrict__ In, float* __restrict__ Out, int total) {
  int i = blockIdx.x * blockDim.x + threadIdx.x;
  if (i >= total) return;
  int pw = i & 31;
  int ph = (i >> 5) & 31;
  long long t = i >> 10;                 // b*32 + c
  const float* p = In + t * 4096 + (ph * 2) * 64 + pw * 2;
  Out[i] = fmaxf(fmaxf(p[0], p[1]), fmaxf(p[64], p[65]));
}

// ---------------------------------------------------------------------------
// Fused non-local attention (flash style, never materializes f[4096,1024]):
//   scores s = theta_tile(16 x 32ic) @ phi_chunk(32ic x 32m)   [WMMA]
//   online softmax over m (row stats live in per-VGPR-row registers)
//   y_acc += P(16 x 32m) @ g_chunk^T(32m x 32ic)               [WMMA]
// phi/g chunk tiles are DMA'd into LDS with global_load_async_to_lds_b32
// (raw f32, fragment-ordered), completed with s_wait_asynccnt.
// Block = 128 thr (4 waves); block tile = 64 query rows; 32-key chunks.
// ---------------------------------------------------------------------------
__global__ void attn_wmma(const float* __restrict__ Theta, const float* __restrict__ PhiP,
                          const float* __restrict__ GP, float* __restrict__ Y) {
  __shared__ alignas(16) _Float16 Th[64 * 32];      // theta^T tile: [n_local][ic] (A frags)
  __shared__ alignas(16) float    PhT[32 * 32];     // phi chunk, frag order [m][ic], f32
  __shared__ alignas(16) float    GcN[32 * 32];     // g chunk,   frag order [ic][m], f32
  __shared__ alignas(16) _Float16 Pb[4][16 * 32];   // per-wave P relayout buffer
  const int tid = threadIdx.x, lane = tid & 31, wv = tid >> 5;
  const int n0 = blockIdx.x * 64, b = blockIdx.z;

  for (int i = tid; i < 64 * 32; i += 128) {
    int nl = i >> 5, ic = i & 31;
    Th[i] = (_Float16)Theta[((long long)b * 32 + ic) * 4096 + n0 + nl];
  }
  __syncthreads();
  v16h aTh = lds_a16(Th + wv * 16 * 32, 32, lane);

  const unsigned ph_base = (unsigned)(size_t)(void*)PhT;
  const unsigned gc_base = (unsigned)(size_t)(void*)GcN;
  const int am  = tid & 31;        // async lane index along m (coalesced)
  const int aic0 = (tid >> 5) * 8; // 8 ic rows per thread

  v8f acc0 = {}, acc1 = {};
  float rm[8], rl[8];
#pragma unroll
  for (int r = 0; r < 8; ++r) { rm[r] = -3.0e30f; rl[r] = 0.0f; }

  for (int m0 = 0; m0 < 1024; m0 += 32) {
    __syncthreads();   // previous chunk's LDS reads done
    // ---- async DMA: phi chunk -> PhT[m][ic], g chunk -> GcN[ic][m] (both f32)
#pragma unroll
    for (int j = 0; j < 8; ++j) {
      int ic = aic0 + j;
      const float* ps = PhiP + ((long long)b * 32 + ic) * 1024 + m0 + am;
      const float* gs = GP   + ((long long)b * 32 + ic) * 1024 + m0 + am;
      async_ld_b32(ph_base + (unsigned)((am * 32 + ic) << 2), ps);
      async_ld_b32(gc_base + (unsigned)((ic * 32 + am) << 2), gs);
    }
    asm volatile("s_wait_asynccnt 0" ::: "memory");
    __syncthreads();

    const int col = lane & 15, kb = lane & 16;
    v16h b0 = lds_frag_f32(PhT + col * 32 + kb);
    v16h b1 = lds_frag_f32(PhT + (16 + col) * 32 + kb);
    v8f s0 = {}, s1 = {};
    s0 = wmma32(aTh, b0, s0);
    s1 = wmma32(aTh, b1, s1);

    // ---- online softmax; D-frag row = r + ((lane&16)>>1), so per-VGPR stats
    float mx[8];
#pragma unroll
    for (int r = 0; r < 8; ++r) mx[r] = fmaxf(s0[r], s1[r]);
#pragma unroll
    for (int off = 1; off < 16; off <<= 1)
#pragma unroll
      for (int r = 0; r < 8; ++r) mx[r] = fmaxf(mx[r], __shfl_xor(mx[r], off, 32));

    float p0[8], p1[8], sc[8], rs[8];
#pragma unroll
    for (int r = 0; r < 8; ++r) {
      float nm = fmaxf(rm[r], mx[r]);
      sc[r] = __expf(rm[r] - nm);
      rm[r] = nm;
      p0[r] = __expf(s0[r] - nm);
      p1[r] = __expf(s1[r] - nm);
      rs[r] = p0[r] + p1[r];
    }
#pragma unroll
    for (int off = 1; off < 16; off <<= 1)
#pragma unroll
      for (int r = 0; r < 8; ++r) rs[r] += __shfl_xor(rs[r], off, 32);
#pragma unroll
    for (int r = 0; r < 8; ++r) {
      rl[r] = rl[r] * sc[r] + rs[r];
      acc0[r] *= sc[r];
      acc1[r] *= sc[r];
    }

    // ---- bounce P through wave-private LDS to get A-fragment layout
    _Float16* pb = Pb[wv];
    const int rh = (lane >> 4) & 1;
#pragma unroll
    for (int r = 0; r < 8; ++r) {
      int row = r + rh * 8;
      pb[row * 32 + col]      = (_Float16)p0[r];
      pb[row * 32 + 16 + col] = (_Float16)p1[r];
    }
    asm volatile("s_wait_dscnt 0" ::: "memory");   // within-wave LDS RAW fence

    v16h aP = lds_a16(pb, 32, lane);
    v16h g0 = lds_frag_f32(GcN + col * 32 + kb);
    v16h g1 = lds_frag_f32(GcN + (16 + col) * 32 + kb);
    acc0 = wmma32(aP, g0, acc0);
    acc1 = wmma32(aP, g1, acc1);
  }

  const int col = lane & 15, rh = (lane >> 4) & 1;
#pragma unroll
  for (int r = 0; r < 8; ++r) {
    int row = r + rh * 8;
    int n = n0 + wv * 16 + row;
    float inv = 1.0f / rl[r];
    Y[((long long)b * 32 + col) * 4096 + n]      = acc0[r] * inv;
    Y[((long long)b * 32 + 16 + col) * 4096 + n] = acc1[r] * inv;
  }
}

// ---------------------------------------------------------------------------
// 3x3 SAME conv as 9 shifted 1x1 GEMMs, OC=32, + bias + ReLU.
// Block handles one output row (64 px) x 32 oc. grid = (H, 1, B).
// In/Out live in the 160-channel feats buffer (channel-prefix concat).
// ---------------------------------------------------------------------------
__global__ void conv3x3_wmma(const float* __restrict__ Wt, const float* __restrict__ In,
                             const float* __restrict__ bias, float* __restrict__ Out,
                             int Cin, long long inb, long long outb) {
  __shared__ alignas(16) _Float16 As[32 * 32];
  __shared__ alignas(16) _Float16 BsT[64 * 32];   // fragment order [n][k]
  const int tid = threadIdx.x, lane = tid & 31, wv = tid >> 5;
  const int h = blockIdx.x, b = blockIdx.z;

  v8f acc0 = {}, acc1 = {};
  for (int kh = 0; kh < 3; ++kh) {
    int ih = h + kh - 1;
    if (ih < 0 || ih >= 64) continue;        // uniform across block
    for (int kw = 0; kw < 3; ++kw) {
      for (int k0 = 0; k0 < Cin; k0 += 32) {
        __syncthreads();
        for (int i = tid; i < 32 * 32; i += 128) {
          int m = i >> 5, kk = i & 31;
          As[i] = (_Float16)Wt[((m * Cin + (k0 + kk)) * 3 + kh) * 3 + kw];
        }
        for (int i = tid; i < 32 * 64; i += 128) {
          int kk = i >> 6, n = i & 63;
          int iw = n + kw - 1;
          float v = (iw >= 0 && iw < 64)
                        ? In[(long long)b * inb + (long long)(k0 + kk) * 4096 + ih * 64 + iw]
                        : 0.0f;
          BsT[n * 32 + kk] = (_Float16)v;
        }
        __syncthreads();
        v16h a0 = lds_a16(As, 32, lane);
        v16h a1 = lds_a16(As + 16 * 32, 32, lane);
        v16h bf = lds_frag_h(BsT + (wv * 16 + (lane & 15)) * 32 + (lane & 16));
        acc0 = wmma32(a0, bf, acc0);
        acc1 = wmma32(a1, bf, acc1);
      }
    }
  }

  const int col = lane & 15, rh = (lane >> 4) & 1;
  const int n = h * 64 + wv * 16 + col;
#pragma unroll
  for (int r = 0; r < 8; ++r) {
    int m0 = r + rh * 8;
    int m1 = m0 + 16;
    Out[(long long)b * outb + (long long)m0 * 4096 + n] = fmaxf(acc0[r] + bias[m0], 0.0f);
    Out[(long long)b * outb + (long long)m1 * 4096 + n] = fmaxf(acc1[r] + bias[m1], 0.0f);
  }
}

// ---------------------------------------------------------------------------
// Host orchestration
// ---------------------------------------------------------------------------
extern "C" void kernel_launch(void* const* d_in, const int* in_sizes, int n_in,
                              void* d_out, int out_size, void* d_ws, size_t ws_size,
                              hipStream_t stream) {
  (void)in_sizes; (void)n_in; (void)out_size; (void)ws_size;
  const float* x       = (const float*)d_in[0];
  const float* g_w     = (const float*)d_in[1];
  const float* g_b     = (const float*)d_in[2];
  const float* theta_w = (const float*)d_in[3];
  const float* theta_b = (const float*)d_in[4];
  const float* phi_w   = (const float*)d_in[5];
  const float* phi_b   = (const float*)d_in[6];
  const float* W_w     = (const float*)d_in[7];
  const float* W_b     = (const float*)d_in[8];
  const float* c0w     = (const float*)d_in[9];
  const float* c0b     = (const float*)d_in[10];
  const float* c1w     = (const float*)d_in[11];
  const float* c1b     = (const float*)d_in[12];
  const float* c2w     = (const float*)d_in[13];
  const float* c2b     = (const float*)d_in[14];
  const float* fw      = (const float*)d_in[15];
  const float* fb      = (const float*)d_in[16];
  float* out = (float*)d_out;

  const int B = 32;
  const long long N = 4096;

  float* ws    = (float*)d_ws;
  float* theta = ws;                                   // B*32*4096
  float* tmpF  = theta + (size_t)B * 32 * 4096;        // B*32*4096 (phi then g, full res)
  float* phiP  = tmpF + (size_t)B * 32 * 4096;         // B*32*1024
  float* gP    = phiP + (size_t)B * 32 * 1024;         // B*32*1024
  float* y     = gP + (size_t)B * 32 * 1024;           // B*32*4096
  float* feats = y + (size_t)B * 32 * 4096;            // B*160*4096

  dim3 blk(128);

  // theta = conv1x1(x, theta_w)
  gemm1x1_wmma<<<dim3(64, 1, B), blk, 0, stream>>>(theta_w, x, theta_b, nullptr, theta,
                                                   64, 64 * N, 0, 32 * N);
  // phi full -> pool
  gemm1x1_wmma<<<dim3(64, 1, B), blk, 0, stream>>>(phi_w, x, phi_b, nullptr, tmpF,
                                                   64, 64 * N, 0, 32 * N);
  maxpool2_k<<<(B * 32 * 1024 + 255) / 256, 256, 0, stream>>>(tmpF, phiP, B * 32 * 1024);
  // g full -> pool (reuse tmpF)
  gemm1x1_wmma<<<dim3(64, 1, B), blk, 0, stream>>>(g_w, x, g_b, nullptr, tmpF,
                                                   64, 64 * N, 0, 32 * N);
  maxpool2_k<<<(B * 32 * 1024 + 255) / 256, 256, 0, stream>>>(tmpF, gP, B * 32 * 1024);

  // fused softmax(theta^T phi) @ g^T  ->  y[b, ic, n]
  attn_wmma<<<dim3(64, 1, B), blk, 0, stream>>>(theta, phiP, gP, y);

  // x_nl = W_w @ y + W_b + x  -> feats channels 0..63
  gemm1x1_wmma<<<dim3(64, 2, B), blk, 0, stream>>>(W_w, y, W_b, x, feats,
                                                   32, 32 * N, 64 * N, 160 * N);

  // dense growth: o0 (ch 64..95), o1 (ch 96..127), o2 (ch 128..159)
  conv3x3_wmma<<<dim3(64, 1, B), blk, 0, stream>>>(c0w, feats, c0b, feats + 64 * 4096,
                                                   64, 160 * N, 160 * N);
  conv3x3_wmma<<<dim3(64, 1, B), blk, 0, stream>>>(c1w, feats, c1b, feats + 96 * 4096,
                                                   96, 160 * N, 160 * N);
  conv3x3_wmma<<<dim3(64, 1, B), blk, 0, stream>>>(c2w, feats, c2b, feats + 128 * 4096,
                                                   128, 160 * N, 160 * N);

  // fusion 1x1 (K=160) + x residual -> out
  gemm1x1_wmma<<<dim3(64, 2, B), blk, 0, stream>>>(fw, feats, fb, x, out,
                                                   160, 160 * N, 64 * N, 64 * N);
}
